// UscLoss_79869211836799
// MI455X (gfx1250) — matrix operand order repaired and checked by reading, not compiled, not run on hardware
//
#include <hip/hip_runtime.h>
#include <hip/hip_bf16.h>
#include <math.h>

typedef __attribute__((ext_vector_type(16))) _Float16 v16h;
typedef __attribute__((ext_vector_type(8)))  _Float16 v8h;
typedef __attribute__((ext_vector_type(8)))  float    v8f;

#define TWO_N 8192
#define HALF_N 4096
#define DDIM 256
// exp(sim / T) = exp2(sim * (1/T) * log2(e)), T = 0.5
#define LOG2E_OVER_T 2.8853900817779268f

// ---------------------------------------------------------------------------
// 1) Row-normalize [emb_i; emb_j] -> fp16 z[8192][256].  One wave per row.
// ---------------------------------------------------------------------------
__global__ void nt_normalize(const float* __restrict__ ei,
                             const float* __restrict__ ej,
                             _Float16* __restrict__ z)
{
    const int row  = blockIdx.x * 8 + (threadIdx.x >> 5);
    const int lane = threadIdx.x & 31;
    const float* src = (row < HALF_N) ? (ei + (size_t)row * DDIM)
                                      : (ej + (size_t)(row - HALF_N) * DDIM);
    float v[8];
    float ss = 0.f;
#pragma unroll
    for (int i = 0; i < 8; ++i) {
        v[i] = src[lane + 32 * i];
        ss += v[i] * v[i];
    }
#pragma unroll
    for (int m = 1; m < 32; m <<= 1) ss += __shfl_xor(ss, m, 32);
    const float nrm = fmaxf(sqrtf(ss), 1e-8f);
    const float inv = 1.0f / nrm;
    _Float16* dst = z + (size_t)row * DDIM;
#pragma unroll
    for (int i = 0; i < 8; ++i) dst[lane + 32 * i] = (_Float16)(v[i] * inv);
}

// ---------------------------------------------------------------------------
// 2) Zero denominator accumulators (workspace is poisoned by harness).
// ---------------------------------------------------------------------------
__global__ void nt_zero(float* __restrict__ p, int n)
{
    const int i = blockIdx.x * blockDim.x + threadIdx.x;
    if (i < n) p[i] = 0.f;
}

// ---------------------------------------------------------------------------
// 3) Positive-pair similarities: pos[k] = z[k] . z[(k+4096) mod 8192].
//    One wave per k; trivial cost, keeps the GEMM epilogue branch-free.
// ---------------------------------------------------------------------------
__global__ void nt_positives(const _Float16* __restrict__ z,
                             float* __restrict__ pos)
{
    const int k    = blockIdx.x * 8 + (threadIdx.x >> 5);
    const int lane = threadIdx.x & 31;
    const int prt  = (k + HALF_N) & (TWO_N - 1);
    const _Float16* a = z + (size_t)k   * DDIM;
    const _Float16* b = z + (size_t)prt * DDIM;
    float s = 0.f;
#pragma unroll
    for (int i = 0; i < 8; ++i)
        s += (float)a[lane + 32 * i] * (float)b[lane + 32 * i];
#pragma unroll
    for (int m = 1; m < 32; m <<= 1) s += __shfl_xor(s, m, 32);
    if (lane == 0) pos[k] = s;
}

// ---------------------------------------------------------------------------
// 4) Fused GEMM (z z^T) + exp row-sum, diagonal masked.
//    grid = (8192/64 row stripes, 8192/512 col stripes), block = 256 (8 waves).
//    Wave w: rows [bx*64 + (w>>1)*16, +16); col tiles (w&1)+2*ct over stripe.
// ---------------------------------------------------------------------------
__global__ void nt_sim_rowsum(const _Float16* __restrict__ z,
                              float* __restrict__ denom)
{
    const int tid  = threadIdx.x;
    const int w    = tid >> 5;       // wave index 0..7
    const int lane = tid & 31;
    const int h    = lane >> 4;      // half-wave: selects K sub-block
    const int n    = lane & 15;      // row-in-tile for A, col-in-tile for B/C

    const int mbase = blockIdx.x * 64 + (w >> 1) * 16;

    // Preload A panel (16 rows x 256 K) in contiguous-K 16x32 A layout (same order as B):
    // lane (h, m=n): element e: K = 16h + e.
    const _Float16* zrow = z + (size_t)(mbase + n) * DDIM;
    v16h A[8];
#pragma unroll
    for (int kt = 0; kt < 8; ++kt) {
        const v8h a0 = *(const v8h*)(zrow + kt * 32 + 16 * h);
        const v8h a1 = *(const v8h*)(zrow + kt * 32 + 16 * h + 8);
#pragma unroll
        for (int e = 0; e < 8; ++e) { A[kt][e] = a0[e]; A[kt][8 + e] = a1[e]; }
    }

    float rowacc[8];
#pragma unroll
    for (int r = 0; r < 8; ++r) rowacc[r] = 0.f;

    for (int ct = 0; ct < 16; ++ct) {
        const int ntile = blockIdx.y * 32 + (w & 1) + 2 * ct;
        const int nbase = ntile * 16;
        // B of z*z^T is rows of z along N: lane (h, n) holds K = 16h + [0..15]
        // of row (nbase+n) -> one contiguous 32-byte load per K-step.
        const _Float16* zcol = z + (size_t)(nbase + n) * DDIM;

        v8f acc = {};
#pragma unroll
        for (int kt = 0; kt < 8; ++kt) {
            const v16h b = *(const v16h*)(zcol + kt * 32 + 16 * h);
            acc = __builtin_amdgcn_wmma_f32_16x16x32_f16(
                false, A[kt], false, b, (short)0, acc, false, false);
        }

        // C/D layout: VGPR r, lane -> M = r + 8h, N = n.
        // Argument of exp2 is bounded (|sim|<=~1, *2.885), so raw v_exp_f32
        // is exact for our range: use the hardware builtin directly.
#pragma unroll
        for (int r = 0; r < 8; ++r) {
            const float s   = acc[r];
            const int  grow = mbase + r + 8 * h;
            const int  gcol = nbase + n;
            const float ex  = __builtin_amdgcn_exp2f(s * LOG2E_OVER_T);
            rowacc[r] += (gcol == grow) ? 0.f : ex;   // mask diagonal
        }
    }

    // Reduce row sums across the 16 lanes of each half-wave, then atomically
    // accumulate the partial denominators (2 atomics per row-VGPR per wave).
#pragma unroll
    for (int r = 0; r < 8; ++r) {
        float v = rowacc[r];
        v += __shfl_xor(v, 1, 32);
        v += __shfl_xor(v, 2, 32);
        v += __shfl_xor(v, 4, 32);
        v += __shfl_xor(v, 8, 32);
        if (n == 0) atomicAdd(&denom[mbase + r + 8 * h], v);
    }
}

// ---------------------------------------------------------------------------
// 5) Final scalar: mean_k( log(denom[k]) - pos[k]/T ).
// ---------------------------------------------------------------------------
__global__ void nt_loss(const float* __restrict__ denom,
                        const float* __restrict__ pos,
                        float* __restrict__ out)
{
    __shared__ float sm[256];
    float p = 0.f;
    for (int k = threadIdx.x; k < TWO_N; k += 256)
        p += logf(denom[k]) - pos[k] * 2.0f;   // pos/T with T = 0.5
    sm[threadIdx.x] = p;
    __syncthreads();
    for (int s = 128; s > 0; s >>= 1) {
        if (threadIdx.x < s) sm[threadIdx.x] += sm[threadIdx.x + s];
        __syncthreads();
    }
    if (threadIdx.x == 0) out[0] = sm[0] / (float)TWO_N;
}

// ---------------------------------------------------------------------------
extern "C" void kernel_launch(void* const* d_in, const int* in_sizes, int n_in,
                              void* d_out, int out_size, void* d_ws, size_t ws_size,
                              hipStream_t stream)
{
    const float* ei = (const float*)d_in[0];   // [4096, 256] fp32
    const float* ej = (const float*)d_in[1];   // [4096, 256] fp32
    float* out = (float*)d_out;                // scalar fp32

    // Workspace layout: z fp16 (4 MB) | denom (32 KB) | pos (32 KB)
    _Float16* z  = (_Float16*)d_ws;
    float* denom = (float*)((char*)d_ws + (size_t)TWO_N * DDIM * sizeof(_Float16));
    float* pos   = denom + TWO_N;

    nt_normalize<<<TWO_N / 8, 256, 0, stream>>>(ei, ej, z);
    nt_zero<<<(TWO_N + 255) / 256, 256, 0, stream>>>(denom, TWO_N);
    nt_positives<<<TWO_N / 8, 256, 0, stream>>>(z, pos);
    nt_sim_rowsum<<<dim3(TWO_N / 64, TWO_N / 512), 256, 0, stream>>>(z, denom);
    nt_loss<<<1, 256, 0, stream>>>(denom, pos, out);
}
